// ThinPlateNetV2_3831110828398
// MI455X (gfx1250) — matrix operand order repaired and checked by reading, not compiled, not run on hardware
//
#include <hip/hip_runtime.h>
#include <hip/hip_bf16.h>

// ---------------------------------------------------------------------------
// ThinPlateNetV2 forward, MI455X (gfx1250, wave32, WMMA bf16 16x16x32)
// All GEMM dimensions are compile-time template parameters so the tap/K loops
// fully unroll into straight-line load+wmma code (no integer division, no
// loop SALU in the hot path).
// ---------------------------------------------------------------------------

typedef __bf16 bf16_t;
typedef __attribute__((ext_vector_type(8)))  __bf16 v8bf;
typedef __attribute__((ext_vector_type(16))) __bf16 v16bf;
typedef __attribute__((ext_vector_type(8)))  float  v8f;

// ----- problem constants ----------------------------------------------------
constexpr int B_   = 4;
constexpr int NKP  = 512;
constexpr int CIN  = 64;
constexpr int HFI  = 128, WFI = 128;     // feature map in
constexpr int HO   = 512, WO  = 512;     // image size
constexpr int HG   = 32,  WG  = 32;      // polar grid
constexpr int C1O  = 128, C1I = 66, C1IP = 96;   // conv1 (K padded to 96)
constexpr int C2O  = 128, C2I = 128;             // conv2
constexpr int HP   = 130, WP  = 130;             // spatially padded input
constexpr int HT   = 126, WT  = 126;             // Theta spatial (VALID)
constexpr int MKP  = B_ * NKP;                   // 2048 keypoints
constexpr int NPAR2 = 132;                       // 66 params * 2
constexpr float BN_S = 0.99999500003749968f;     // 1/sqrt(1+1e-5)
constexpr float KF   = 0.3f;
constexpr float OFFS = 0.35f;                    // (1-KF)/2
constexpr float MAXR = 32.0f;

// ----- workspace layout -----------------------------------------------------
constexpr size_t algn(size_t x) { return (x + 255) & ~size_t(255); }
constexpr size_t OFF_XP  = 0;
constexpr size_t SZ_XP   = (size_t)B_ * HP * WP * C1IP * 2;
constexpr size_t OFF_W1T = algn(OFF_XP + SZ_XP);
constexpr size_t SZ_W1T  = (size_t)9 * C1O * C1IP * 2;
constexpr size_t OFF_W2T = algn(OFF_W1T + SZ_W1T);
constexpr size_t SZ_W2T  = (size_t)9 * C2O * C2I * 2;
constexpr size_t OFF_L1W = algn(OFF_W2T + SZ_W2T);
constexpr size_t SZ_L1W  = (size_t)256 * 128 * 2;
constexpr size_t OFF_L2W = algn(OFF_L1W + SZ_L1W);
constexpr size_t SZ_L2W  = (size_t)256 * 256 * 2;
constexpr size_t OFF_L3W = algn(OFF_L2W + SZ_L2W);
constexpr size_t SZ_L3W  = (size_t)NPAR2 * 256 * 2;
constexpr size_t OFF_H1  = algn(OFF_L3W + SZ_L3W);
constexpr size_t SZ_H1   = (size_t)B_ * HFI * WFI * C1O * 2;      // bf16 NHWC
constexpr size_t OFF_TH  = algn(OFF_H1 + SZ_H1);
constexpr size_t SZ_TH   = (size_t)B_ * HT * WT * C2O * 4;        // fp32 NHWC
constexpr size_t OFF_THB = algn(OFF_TH + SZ_TH);
constexpr size_t SZ_THB  = (size_t)MKP * 128 * 2;
constexpr size_t OFF_F1  = algn(OFF_THB + SZ_THB);
constexpr size_t SZ_F1   = (size_t)MKP * 256 * 2;
constexpr size_t OFF_F2  = algn(OFF_F1 + SZ_F1);
constexpr size_t SZ_F2   = (size_t)MKP * 256 * 2;
constexpr size_t OFF_THP = algn(OFF_F2 + SZ_F2);
constexpr size_t SZ_THP  = (size_t)MKP * NPAR2 * 4;
constexpr size_t OFF_WRP = algn(OFF_THP + SZ_THP);

// ----- helpers --------------------------------------------------------------
__device__ __forceinline__ unsigned short f2bf(float f) {
  unsigned u = __float_as_uint(f);
  unsigned r = u + 0x7FFFu + ((u >> 16) & 1u);          // round-to-nearest-even
  return (unsigned short)(r >> 16);
}

__device__ __forceinline__ v16bf load_frag16(const bf16_t* p0, const bf16_t* p1) {
  v8bf lo = *(const v8bf*)p0;
  v8bf hi = *(const v8bf*)p1;
  return __builtin_shufflevector(lo, hi, 0,1,2,3,4,5,6,7,8,9,10,11,12,13,14,15);
}

// ===========================================================================
// prep kernels: bf16 conversion / layout
// ===========================================================================

// x (NCHW f32) + analytic coord channels -> padded NHWC bf16, C padded to 96
__global__ void prep_x_kernel(const float* __restrict__ x,
                              unsigned short* __restrict__ xp) {
  int idx = blockIdx.x * 256 + threadIdx.x;
  int total = B_ * HP * WP * C1IP;
  if (idx >= total) return;
  int c = idx % C1IP; int r = idx / C1IP;
  int xx = r % WP; r /= WP; int yy = r % HP; int b = r / HP;
  float val = 0.0f;
  if (xx >= 1 && xx <= WFI && yy >= 1 && yy <= HFI && c < C1I) {
    int xi = xx - 1, yi = yy - 1;
    if (c < CIN)        val = x[(((size_t)b * CIN + c) * HFI + yi) * WFI + xi];
    else if (c == CIN)  val = -1.0f + 2.0f * xi / (float)(WFI - 1);
    else                val = -1.0f + 2.0f * yi / (float)(HFI - 1);
  }
  xp[idx] = f2bf(val);
}

// conv weight OIHW f32 -> [tap][O][Cpad] bf16 (BN scale folded)
__global__ void prep_cw_kernel(const float* __restrict__ w,
                               unsigned short* __restrict__ wt,
                               int Cout, int Cin, int Cp, float scale) {
  int idx = blockIdx.x * 256 + threadIdx.x;
  int total = 9 * Cout * Cp;
  if (idx >= total) return;
  int c = idx % Cp; int r = idx / Cp; int o = r % Cout; int tap = r / Cout;
  float val = 0.0f;
  if (c < Cin)
    val = w[(((size_t)o * Cin + c) * 3 + tap / 3) * 3 + (tap % 3)] * scale;
  wt[idx] = f2bf(val);
}

__global__ void cvt_bf16_kernel(const float* __restrict__ src,
                                unsigned short* __restrict__ dst,
                                int n, float scale) {
  int i = blockIdx.x * 256 + threadIdx.x;
  if (i < n) dst[i] = f2bf(src[i] * scale);
}

// ===========================================================================
// implicit-GEMM conv (sum over 9 taps), NHWC bf16 in, ReLU epilogue
// block: 256 thr = 8 waves; tile M=64 pixels x N=COUT(128); wave tile 32x32
// All dims compile-time: tap/K loops fully unroll -> straight-line wmma code.
// ===========================================================================
template <bool OUT_BF16, int HIN, int WIN, int CP, int HOUT, int WOUT, int COUT>
__global__ __launch_bounds__(256)
void conv_wmma_kernel(const bf16_t* __restrict__ in, const bf16_t* __restrict__ wt,
                      void* __restrict__ outp) {
  constexpr int PTOT = B_ * HOUT * WOUT;
  const int tid  = threadIdx.x;
  const int lane = tid & 31, wid = tid >> 5;
  const int wm = wid >> 2;                 // 0..1 -> M offset 32*wm
  const int wn = wid & 3;                  // 0..3 -> N offset 32*wn
  const int p0 = blockIdx.x * 64;
  const int klane = (lane >> 4) << 3;      // A K-chunk select: 0 or 8

  int aoff[2];
#pragma unroll
  for (int ms = 0; ms < 2; ++ms) {
    int p = p0 + wm * 32 + ms * 16 + (lane & 15);
    if (p >= PTOT) p = PTOT - 1;
    int ox = p % WOUT; int t = p / WOUT; int oy = t % HOUT; int b = t / HOUT;
    aoff[ms] = ((b * HIN + oy) * WIN + ox) * CP;
  }
  const bf16_t* bbase[2];
#pragma unroll
  for (int ns = 0; ns < 2; ++ns) {
    int n = wn * 32 + ns * 16 + (lane & 15);
    bbase[ns] = wt + (size_t)n * CP + ((lane >> 4) << 3);
  }

  v8f acc[2][2] = {};
#pragma unroll
  for (int tap = 0; tap < 9; ++tap) {
    const int toff = ((tap / 3) * WIN + (tap % 3)) * CP;
    const size_t wtap = (size_t)tap * COUT * CP;
#pragma unroll
    for (int k0 = 0; k0 < CP; k0 += 32) {
      v16bf afr[2], bfr[2];
#pragma unroll
      for (int ms = 0; ms < 2; ++ms) {
        const bf16_t* pa = in + aoff[ms] + toff + k0 + klane;
        afr[ms] = load_frag16(pa, pa + 16);
      }
#pragma unroll
      for (int ns = 0; ns < 2; ++ns) {
        const bf16_t* pb = bbase[ns] + wtap + k0;
        bfr[ns] = load_frag16(pb, pb + 16);
      }
#pragma unroll
      for (int ms = 0; ms < 2; ++ms)
#pragma unroll
        for (int ns = 0; ns < 2; ++ns)
          acc[ms][ns] = __builtin_amdgcn_wmma_f32_16x16x32_bf16(
              false, afr[ms], false, bfr[ns], (short)0, acc[ms][ns],
              false, false);
    }
  }

#pragma unroll
  for (int ms = 0; ms < 2; ++ms)
#pragma unroll
    for (int ns = 0; ns < 2; ++ns) {
      int n = wn * 32 + ns * 16 + (lane & 15);
#pragma unroll
      for (int v = 0; v < 8; ++v) {
        int p = p0 + wm * 32 + ms * 16 + v + ((lane >> 4) << 3);
        if (p < PTOT) {
          float val = fmaxf(acc[ms][ns][v], 0.0f);
          size_t idx = (size_t)p * COUT + n;
          if constexpr (OUT_BF16) ((unsigned short*)outp)[idx] = f2bf(val);
          else                    ((float*)outp)[idx] = val;
        }
      }
    }
}

// ===========================================================================
// dense GEMM: D = act(A[MxK] @ Wt[NxK]^T + bias*bscale)
// block: 256 thr = 8 waves; tile M=64 x N=64; wave tile 16x32
// ACT: 0 = relu, 1 = tanh.  N,K compile-time -> K loop fully unrolls.
// ===========================================================================
template <bool OUT_BF16, int ACT, int N, int K>
__global__ __launch_bounds__(256)
void gemm_wmma_kernel(const bf16_t* __restrict__ A, const bf16_t* __restrict__ Wt,
                      const float* __restrict__ bias, float bscale,
                      void* __restrict__ outp, int M) {
  const int tid  = threadIdx.x;
  const int lane = tid & 31, wid = tid >> 5;
  const int wm = wid >> 1;                         // 0..3
  const int wn = wid & 1;                          // 0..1
  const int m0 = blockIdx.x * 64 + wm * 16;
  const int n0 = blockIdx.y * 64 + wn * 32;
  const int klane = (lane >> 4) << 3;

  int mrow = m0 + (lane & 15);
  if (mrow >= M) mrow = M - 1;
  const bf16_t* arow = A + (size_t)mrow * K + klane;

  const bf16_t* brow[2];
#pragma unroll
  for (int ns = 0; ns < 2; ++ns) {
    int n = n0 + ns * 16 + (lane & 15);
    if (n >= N) n = N - 1;
    brow[ns] = Wt + (size_t)n * K + ((lane >> 4) << 3);
  }

  v8f acc[2] = {};
#pragma unroll
  for (int k0 = 0; k0 < K; k0 += 32) {
    v16bf af = load_frag16(arow + k0, arow + k0 + 16);
#pragma unroll
    for (int ns = 0; ns < 2; ++ns) {
      v16bf bf = load_frag16(brow[ns] + k0, brow[ns] + k0 + 16);
      acc[ns] = __builtin_amdgcn_wmma_f32_16x16x32_bf16(
          false, af, false, bf, (short)0, acc[ns], false, false);
    }
  }

#pragma unroll
  for (int ns = 0; ns < 2; ++ns) {
    int n = n0 + ns * 16 + (lane & 15);
    if (n >= N) continue;
    float bv = bias[n] * bscale;
#pragma unroll
    for (int v = 0; v < 8; ++v) {
      int m = m0 + v + ((lane >> 4) << 3);
      if (m < M) {
        float val = acc[ns][v] + bv;
        if constexpr (ACT == 0) val = fmaxf(val, 0.0f);
        else                    val = tanhf(val);
        size_t idx = (size_t)m * N + n;
        if constexpr (OUT_BF16) ((unsigned short*)outp)[idx] = f2bf(val);
        else                    ((float*)outp)[idx] = val;
      }
    }
  }
}

// ===========================================================================
// bilinear gather of Theta (fp32 NHWC) at keypoints -> bf16 [2048][128]
// ===========================================================================
__global__ __launch_bounds__(128)
void sample_theta_kernel(const float* __restrict__ Theta,
                         const int* __restrict__ keypts,
                         unsigned short* __restrict__ thb) {
  int m = blockIdx.x;                 // b*512 + k
  int c = threadIdx.x;                // channel
  int b = m >> 9;
  float fx = (float)keypts[m * 2 + 0] * (float)(WT - 1) / (float)(WO - 1);
  float fy = (float)keypts[m * 2 + 1] * (float)(HT - 1) / (float)(HO - 1);
  float x0 = floorf(fx), y0 = floorf(fy);
  float wx = fx - x0, wy = fy - y0;
  float acc = 0.0f;
#pragma unroll
  for (int ty = 0; ty < 2; ++ty)
#pragma unroll
    for (int tx = 0; tx < 2; ++tx) {
      int xi = (int)x0 + tx, yi = (int)y0 + ty;
      float wgt = (tx ? wx : 1.0f - wx) * (ty ? wy : 1.0f - wy);
      if (xi >= 0 && xi < WT && yi >= 0 && yi < HT)
        acc += Theta[(((size_t)b * HT + yi) * WT + xi) * C2O + c] * wgt;
    }
  thb[(size_t)m * 128 + c] = f2bf(acc);
}

// ===========================================================================
// TPS warp: polar grid, min/max, ctrl points, RBF warp (per keypoint block)
// ===========================================================================
__device__ __forceinline__ float blk_red(float v, bool mn, float* s, int tid) {
  s[tid] = v; __syncthreads();
  for (int off = 128; off > 0; off >>= 1) {
    if (tid < off) s[tid] = mn ? fminf(s[tid], s[tid + off])
                               : fmaxf(s[tid], s[tid + off]);
    __syncthreads();
  }
  float r = s[0]; __syncthreads();
  return r;
}

__global__ __launch_bounds__(256)
void tps_warp_kernel(const float* __restrict__ theta,
                     const int* __restrict__ keypts,
                     float* __restrict__ warp) {
  __shared__ float s_pgn[HG * WG][2];
  __shared__ float s_red[256];
  __shared__ float s_ctrl[64][2];
  __shared__ float s_waug[64][2];
  __shared__ float s_a[3][2];

  const int m = blockIdx.x, tid = threadIdx.x;
  const float xk = (float)keypts[m * 2 + 0] / (float)WO * 2.0f - 1.0f;
  const float yk = (float)keypts[m * 2 + 1] / (float)HO * 2.0f - 1.0f;

  float mnx = 3e38f, mny = 3e38f, mxx = -3e38f, mxy = -3e38f;
#pragma unroll
  for (int j = 0; j < 4; ++j) {
    int i = tid + j * 256;
    int h = i >> 5, w = i & 31;
    float xs = ((float)w + 0.5f) * (2.0f / WG) - 1.0f;
    float ys = ((float)h + 0.5f) * (2.0f / HG) - 1.0f;
    float r_  = 1.0f + ((xs + 1.0f) * 0.5f) * (MAXR - 1.0f);
    float r_s = (r_ - 1.0f) / (MAXR - 1.0f) * 2.0f * MAXR / (float)WO;
    float t_s = (ys + 1.0f) * 3.14159265358979323846f;
    float x_s = r_s * cosf(t_s) + xk;
    float y_s = r_s * sinf(t_s) * ((float)WO / (float)HO) + yk;
    s_pgn[i][0] = x_s; s_pgn[i][1] = y_s;
    mnx = fminf(mnx, x_s); mxx = fmaxf(mxx, x_s);
    mny = fminf(mny, y_s); mxy = fmaxf(mxy, y_s);
  }
  float vminx = blk_red(mnx, true,  s_red, tid);
  float vmaxx = blk_red(mxx, false, s_red, tid);
  float vminy = blk_red(mny, true,  s_red, tid);
  float vmaxy = blk_red(mxy, false, s_red, tid);
  float ptpx = vmaxx - vminx, ptpy = vmaxy - vminy;

#pragma unroll
  for (int j = 0; j < 4; ++j) {
    int i = tid + j * 256;
    s_pgn[i][0] = (s_pgn[i][0] - vminx) / (ptpx + 1e-8f) * KF + OFFS;
    s_pgn[i][1] = (s_pgn[i][1] - vminy) / (ptpy + 1e-8f) * KF + OFFS;
  }
  __syncthreads();

  if (tid < 64) {
    int r = (tid >> 3) * 4, c = (tid & 7) * 4;       // 8x8 ctrl, stride 4
    s_ctrl[tid][0] = s_pgn[r * WG + c][0];
    s_ctrl[tid][1] = s_pgn[r * WG + c][1];
  }
  if (tid < 2) {                                      // augmented TPS weights
    const float* th = theta + (size_t)m * NPAR2;      // [66][2]
    float sum = 0.0f;
    for (int t = 0; t < 63; ++t) {
      float v = th[t * 2 + tid];
      s_waug[t + 1][tid] = v;
      sum += v;
    }
    s_waug[0][tid] = -sum;
  }
  if (tid < 6) {                                      // affine part (3x2)
    int r = tid >> 1, ch = tid & 1;
    s_a[r][ch] = theta[(size_t)m * NPAR2 + (63 + r) * 2 + ch];
  }
  __syncthreads();

#pragma unroll
  for (int j = 0; j < 4; ++j) {
    int i = tid + j * 256;
    float x = s_pgn[i][0], y = s_pgn[i][1];
    float bx = 0.0f, by = 0.0f;
    for (int t = 0; t < 64; ++t) {
      float dx = x - s_ctrl[t][0], dy = y - s_ctrl[t][1];
      float r2 = fmaxf(dx * dx + dy * dy, 0.0f);
      float U = r2 * logf(sqrtf(r2 + 1e-12f) + 1e-6f);
      bx += U * s_waug[t][0];
      by += U * s_waug[t][1];
    }
    float zx = s_a[0][0] + x * s_a[1][0] + y * s_a[2][0];
    float zy = s_a[0][1] + x * s_a[1][1] + y * s_a[2][1];
    float wxv = x + zx + bx, wyv = y + zy + by;
    warp[((size_t)m * 1024 + i) * 2 + 0] = (wxv - OFFS) / KF * ptpx + vminx;
    warp[((size_t)m * 1024 + i) * 2 + 1] = (wyv - OFFS) / KF * ptpy + vminy;
  }
}

// ===========================================================================
// final bilinear resample of in_imgs -> out (B,512,3,32,32)
// ===========================================================================
__global__ __launch_bounds__(256)
void sample_out_kernel(const float* __restrict__ imgs,
                       const float* __restrict__ warp,
                       float* __restrict__ out) {
  const int m = blockIdx.x, tid = threadIdx.x;
  const int b = m >> 9;
#pragma unroll
  for (int j = 0; j < 4; ++j) {
    int i = tid + j * 256;
    float gx = warp[((size_t)m * 1024 + i) * 2 + 0];
    float gy = warp[((size_t)m * 1024 + i) * 2 + 1];
    float ix = ((gx + 1.0f) * (float)WO - 1.0f) * 0.5f;
    float iy = ((gy + 1.0f) * (float)HO - 1.0f) * 0.5f;
    float x0 = floorf(ix), y0 = floorf(iy);
    float wx = ix - x0, wy = iy - y0;
    float c0 = 0.0f, c1 = 0.0f, c2 = 0.0f;
#pragma unroll
    for (int ty = 0; ty < 2; ++ty)
#pragma unroll
      for (int tx = 0; tx < 2; ++tx) {
        int xi = (int)x0 + tx, yi = (int)y0 + ty;
        float wgt = (tx ? wx : 1.0f - wx) * (ty ? wy : 1.0f - wy);
        if (xi >= 0 && xi < WO && yi >= 0 && yi < HO) {
          size_t base = (((size_t)b * 3) * HO + yi) * WO + xi;
          c0 += imgs[base] * wgt;
          c1 += imgs[base + (size_t)HO * WO] * wgt;
          c2 += imgs[base + 2 * (size_t)HO * WO] * wgt;
        }
      }
    size_t ob = ((size_t)m * 3) * 1024 + i;
    out[ob]        = c0;
    out[ob + 1024] = c1;
    out[ob + 2048] = c2;
  }
}

// ===========================================================================
// launch
// ===========================================================================
extern "C" void kernel_launch(void* const* d_in, const int* in_sizes, int n_in,
                              void* d_out, int out_size, void* d_ws, size_t ws_size,
                              hipStream_t stream) {
  (void)in_sizes; (void)n_in; (void)out_size; (void)ws_size;
  const float* x    = (const float*)d_in[0];
  const float* imgs = (const float*)d_in[1];
  const int*   kp   = (const int*)d_in[2];
  const float* w1   = (const float*)d_in[3];
  const float* w2   = (const float*)d_in[4];
  const float* l1w  = (const float*)d_in[5];
  const float* l1b  = (const float*)d_in[6];
  const float* l2w  = (const float*)d_in[7];
  const float* l2b  = (const float*)d_in[8];
  const float* l3w  = (const float*)d_in[9];
  const float* l3b  = (const float*)d_in[10];

  char* ws = (char*)d_ws;
  bf16_t*         XP   = (bf16_t*)(ws + OFF_XP);
  unsigned short* XPu  = (unsigned short*)(ws + OFF_XP);
  bf16_t*         W1T  = (bf16_t*)(ws + OFF_W1T);
  unsigned short* W1Tu = (unsigned short*)(ws + OFF_W1T);
  bf16_t*         W2T  = (bf16_t*)(ws + OFF_W2T);
  unsigned short* W2Tu = (unsigned short*)(ws + OFF_W2T);
  bf16_t*         L1W  = (bf16_t*)(ws + OFF_L1W);
  unsigned short* L1Wu = (unsigned short*)(ws + OFF_L1W);
  bf16_t*         L2W  = (bf16_t*)(ws + OFF_L2W);
  unsigned short* L2Wu = (unsigned short*)(ws + OFF_L2W);
  bf16_t*         L3W  = (bf16_t*)(ws + OFF_L3W);
  unsigned short* L3Wu = (unsigned short*)(ws + OFF_L3W);
  bf16_t*         H1   = (bf16_t*)(ws + OFF_H1);
  float*          THf  = (float*)(ws + OFF_TH);
  bf16_t*         THB  = (bf16_t*)(ws + OFF_THB);
  unsigned short* THBu = (unsigned short*)(ws + OFF_THB);
  bf16_t*         F1   = (bf16_t*)(ws + OFF_F1);
  bf16_t*         F2   = (bf16_t*)(ws + OFF_F2);
  float*          THPf = (float*)(ws + OFF_THP);
  float*          WRPf = (float*)(ws + OFF_WRP);

  // --- prep / bf16 conversion (BN scale folded into conv & lin1 weights) ---
  {
    int total = B_ * HP * WP * C1IP;
    prep_x_kernel<<<(total + 255) / 256, 256, 0, stream>>>(x, XPu);
  }
  prep_cw_kernel<<<(9 * C1O * C1IP + 255) / 256, 256, 0, stream>>>(
      w1, W1Tu, C1O, C1I, C1IP, BN_S);
  prep_cw_kernel<<<(9 * C2O * C2I + 255) / 256, 256, 0, stream>>>(
      w2, W2Tu, C2O, C2I, C2I, BN_S);
  cvt_bf16_kernel<<<(256 * 128 + 255) / 256, 256, 0, stream>>>(l1w, L1Wu, 256 * 128, BN_S);
  cvt_bf16_kernel<<<(256 * 256 + 255) / 256, 256, 0, stream>>>(l2w, L2Wu, 256 * 256, 1.0f);
  cvt_bf16_kernel<<<(NPAR2 * 256 + 255) / 256, 256, 0, stream>>>(l3w, L3Wu, NPAR2 * 256, 1.0f);

  // --- conv1 (SAME, 66->128, padded K=96) : WMMA implicit GEMM ---
  conv_wmma_kernel<true, HP, WP, C1IP, HFI, WFI, C1O>
      <<<(B_ * HFI * WFI) / 64, 256, 0, stream>>>(XP, W1T, (void*)H1);

  // --- conv2 (VALID, 128->128) : WMMA implicit GEMM, fp32 out ---
  conv_wmma_kernel<false, HFI, WFI, C2I, HT, WT, C2O>
      <<<(B_ * HT * WT + 63) / 64, 256, 0, stream>>>(H1, W2T, (void*)THf);

  // --- gather keypoint features ---
  sample_theta_kernel<<<MKP, 128, 0, stream>>>(THf, kp, THBu);

  // --- MLP head (WMMA) ---
  gemm_wmma_kernel<true, 0, 256, 128><<<dim3(MKP / 64, 4), 256, 0, stream>>>(
      THB, L1W, l1b, BN_S, (void*)F1, MKP);
  gemm_wmma_kernel<true, 0, 256, 256><<<dim3(MKP / 64, 4), 256, 0, stream>>>(
      F1, L2W, l2b, 1.0f, (void*)F2, MKP);
  gemm_wmma_kernel<false, 1, NPAR2, 256><<<dim3(MKP / 64, 3), 256, 0, stream>>>(
      F2, L3W, l3b, 1.0f, (void*)THPf, MKP);

  // --- TPS warp + final resample ---
  tps_warp_kernel<<<MKP, 256, 0, stream>>>(THPf, kp, WRPf);
  sample_out_kernel<<<MKP, 256, 0, stream>>>(imgs, WRPf, (float*)d_out);
}
